// PhaseAttentionHybrid_84894323572826
// MI455X (gfx1250) — matrix operand order repaired and checked by reading, not compiled, not run on hardware
//
#include <hip/hip_runtime.h>

typedef __bf16 bf16_t;
typedef __attribute__((ext_vector_type(16))) __bf16 v16bf;
typedef __attribute__((ext_vector_type(8)))  __bf16 v8bf;
typedef __attribute__((ext_vector_type(8)))  float  v8f;

union ABfrag { v16bf v; v8bf h[2]; };
union CFrag  { v8f v; float f[8]; };

constexpr int kB  = 2;
constexpr int kS  = 2048;
constexpr int kD  = 2048;
constexpr int kNH = 16;
constexpr int kHD = 128;
constexpr int kM  = kB * kS;   // 4096 rows

#define SWZ_F(v, imm) __int_as_float(__builtin_amdgcn_ds_swizzle(__float_as_int(v), (imm)))

__device__ __forceinline__ float rowred_max16(float v) {
  v = fmaxf(v, SWZ_F(v, 0x041f));
  v = fmaxf(v, SWZ_F(v, 0x081f));
  v = fmaxf(v, SWZ_F(v, 0x101f));
  v = fmaxf(v, SWZ_F(v, 0x201f));
  return v;
}
__device__ __forceinline__ float rowred_sum16(float v) {
  v += SWZ_F(v, 0x041f);
  v += SWZ_F(v, 0x081f);
  v += SWZ_F(v, 0x101f);
  v += SWZ_F(v, 0x201f);
  return v;
}

__device__ __forceinline__ v8f wmma_bf16(v16bf a, v16bf b, v8f c) {
  return __builtin_amdgcn_wmma_f32_16x16x32_bf16(false, a, false, b, (short)0, c,
                                                 false, false);
}

// A-fragment: 16 contiguous bf16 (contiguous k order, matches B)
__device__ __forceinline__ void load_afrag(ABfrag& f, const bf16_t* p) {
  f.h[0] = *reinterpret_cast<const v8bf*>(p);
  f.h[1] = *reinterpret_cast<const v8bf*>(p + 8);
}
// B-fragment: 16 contiguous bf16 (contiguous k order, matches A, lane = column)
__device__ __forceinline__ void load_bfrag(ABfrag& f, const bf16_t* p) {
  f.h[0] = *reinterpret_cast<const v8bf*>(p);
  f.h[1] = *reinterpret_cast<const v8bf*>(p + 8);
}

// ---------------------------------------------------------------- fp32 -> bf16
__global__ void k_cvt_x(const float* __restrict__ x, bf16_t* __restrict__ xb) {
  int i = blockIdx.x * blockDim.x + threadIdx.x;   // one 8-elem pack per thread
  size_t base = (size_t)i * 8;
  float4 a = *reinterpret_cast<const float4*>(x + base);
  float4 b = *reinterpret_cast<const float4*>(x + base + 4);
  v8bf o;
  o[0] = (bf16_t)a.x; o[1] = (bf16_t)a.y; o[2] = (bf16_t)a.z; o[3] = (bf16_t)a.w;
  o[4] = (bf16_t)b.x; o[5] = (bf16_t)b.y; o[6] = (bf16_t)b.z; o[7] = (bf16_t)b.w;
  *reinterpret_cast<v8bf*>(xb + base) = o;
}

// ------------------------------------------- W [K][N] fp32 -> Wt [N][K] bf16
__global__ void k_wt(const float* __restrict__ W, bf16_t* __restrict__ Wt) {
  __shared__ float tile[32][33];
  int tx = threadIdx.x & 31, ty = threadIdx.x >> 5;   // 32 x 8
  int bx = blockIdx.x * 32, by = blockIdx.y * 32;
#pragma unroll
  for (int i = 0; i < 4; ++i)
    tile[ty + i * 8][tx] = W[(size_t)(by + ty + i * 8) * kD + bx + tx];
  __syncthreads();
#pragma unroll
  for (int i = 0; i < 4; ++i)
    Wt[(size_t)(bx + ty + i * 8) * kD + by + tx] = (bf16_t)tile[tx][ty + i * 8];
}

// -------------------------------------------------- phase path (fp32 scalar)
__global__ void k_phase1(const float* __restrict__ x,
                         const float* __restrict__ Wpq, const float* __restrict__ Wpk,
                         float* __restrict__ cpq, float* __restrict__ spq,
                         float* __restrict__ cpk, float* __restrict__ spk) {
  __shared__ float red[256][32];
  int row = blockIdx.x;                      // [0, 4096)
  const float* xr = x + (size_t)row * kD;
  float accq[kNH], acck[kNH];
#pragma unroll
  for (int h = 0; h < kNH; ++h) { accq[h] = 0.f; acck[h] = 0.f; }
  for (int d = threadIdx.x; d < kD; d += 256) {
    float xv = xr[d];
#pragma unroll
    for (int h = 0; h < kNH; ++h) {
      accq[h] += xv * Wpq[d * kNH + h];
      acck[h] += xv * Wpk[d * kNH + h];
    }
  }
#pragma unroll
  for (int h = 0; h < kNH; ++h) {
    red[threadIdx.x][h]       = accq[h];
    red[threadIdx.x][h + kNH] = acck[h];
  }
  __syncthreads();
  if (threadIdx.x < 32) {
    int col = threadIdx.x;
    float s = 0.f;
    for (int t = 0; t < 256; ++t) s += red[t][col];
    if (col < kNH) {
      cpq[row * kNH + col] = __cosf(s);
      spq[row * kNH + col] = __sinf(s);
    } else {
      cpk[row * kNH + col - kNH] = __cosf(s);
      spk[row * kNH + col - kNH] = __sinf(s);
    }
  }
}

__global__ void k_phase2(const float* __restrict__ cpk, const float* __restrict__ spk,
                         float* __restrict__ Ck, float* __restrict__ Sk) {
  __shared__ float rc[256], rs[256];
  int bh = blockIdx.x;                 // b*16 + h
  int b = bh >> 4, h = bh & 15;
  float sc = 0.f, ss = 0.f;
  for (int s = threadIdx.x; s < kS; s += 256) {
    int idx = (b * kS + s) * kNH + h;
    sc += cpk[idx];
    ss += spk[idx];
  }
  rc[threadIdx.x] = sc; rs[threadIdx.x] = ss;
  __syncthreads();
  for (int off = 128; off > 0; off >>= 1) {
    if (threadIdx.x < off) {
      rc[threadIdx.x] += rc[threadIdx.x + off];
      rs[threadIdx.x] += rs[threadIdx.x + off];
    }
    __syncthreads();
  }
  if (threadIdx.x == 0) { Ck[bh] = rc[0]; Sk[bh] = rs[0]; }
}

__global__ void k_phase3(const float* __restrict__ cpq, const float* __restrict__ spq,
                         const float* __restrict__ Ck, const float* __restrict__ Sk,
                         float* __restrict__ phs) {
  int i = blockIdx.x * blockDim.x + threadIdx.x;   // [0, 4096)
  int b = i >> 11;
  float acc = 0.f;
#pragma unroll
  for (int h = 0; h < kNH; ++h)
    acc += cpq[i * kNH + h] * Ck[b * kNH + h] + spq[i * kNH + h] * Sk[b * kNH + h];
  phs[i] = acc / (float)(kS * kNH);
}

// ------------------------------------------------------------ QKV projection
// One wave computes a 32x64 tile. k-loop manually unrolled by 2 with named
// ping/pong A fragments (a0/a1) -> all array indices are compile-time
// constants, so everything stays in VGPRs (no SROA demotion to LDS/scratch).
__global__ void __launch_bounds__(256)
k_gemm_qkv(const bf16_t* __restrict__ xb,
           const bf16_t* __restrict__ Wqt, const bf16_t* __restrict__ Wkt,
           const bf16_t* __restrict__ Wvt,
           bf16_t* __restrict__ qb, bf16_t* __restrict__ kb, bf16_t* __restrict__ vt) {
  int wid  = (blockIdx.x * blockDim.x + threadIdx.x) >> 5;
  int lane = threadIdx.x & 31;
  int ntile = wid & 31;
  int t     = wid >> 5;
  int mtile = t & 127;
  int mat   = t >> 7;                      // 0:q 1:k 2:v
  const bf16_t* Wt = (mat == 0) ? Wqt : (mat == 1 ? Wkt : Wvt);
  int m0 = mtile * 32, n0 = ntile * 64;
  int rA = lane & 15;
  int aoff = (lane < 16) ? 0 : 16;
  int boff = (lane < 16) ? 0 : 16;
  const bf16_t* arow0 = xb + (size_t)(m0 + rA) * kD + aoff;
  const bf16_t* arow1 = xb + (size_t)(m0 + 16 + rA) * kD + aoff;
  const bf16_t* brow  = Wt + (size_t)(n0 + rA) * kD + boff;   // +c*16*kD per col grp

  CFrag acc[2][4];
#pragma unroll
  for (int mi = 0; mi < 2; ++mi)
#pragma unroll
    for (int c = 0; c < 4; ++c)
#pragma unroll
      for (int r = 0; r < 8; ++r) acc[mi][c].f[r] = 0.f;

  ABfrag a0[2], a1[2], b[4];
  load_afrag(a0[0], arow0);
  load_afrag(a0[1], arow1);

  for (int k = 0; k < kD; k += 64) {
    // ---- half-step 0: consume a0 @ k, prefetch a1 @ k+32
#pragma unroll
    for (int c = 0; c < 4; ++c) load_bfrag(b[c], brow + (size_t)c * 16 * kD + k);
    load_afrag(a1[0], arow0 + k + 32);
    load_afrag(a1[1], arow1 + k + 32);
#pragma unroll
    for (int c = 0; c < 4; ++c) {
      acc[0][c].v = wmma_bf16(a0[0].v, b[c].v, acc[0][c].v);
      acc[1][c].v = wmma_bf16(a0[1].v, b[c].v, acc[1][c].v);
    }
    // ---- half-step 1: consume a1 @ k+32, prefetch a0 @ k+64
#pragma unroll
    for (int c = 0; c < 4; ++c) load_bfrag(b[c], brow + (size_t)c * 16 * kD + k + 32);
    if (k + 64 < kD) {
      load_afrag(a0[0], arow0 + k + 64);
      load_afrag(a0[1], arow1 + k + 64);
    }
#pragma unroll
    for (int c = 0; c < 4; ++c) {
      acc[0][c].v = wmma_bf16(a1[0].v, b[c].v, acc[0][c].v);
      acc[1][c].v = wmma_bf16(a1[1].v, b[c].v, acc[1][c].v);
    }
  }

  int rowadd = (lane < 16) ? 0 : 8;
#pragma unroll
  for (int mi = 0; mi < 2; ++mi) {
#pragma unroll
    for (int c = 0; c < 4; ++c) {
      int n = n0 + c * 16 + rA;
      int h = n >> 7, hd = n & 127;
#pragma unroll
      for (int r = 0; r < 8; ++r) {
        int m = m0 + mi * 16 + r + rowadd;
        int bb = m >> 11, s = m & 2047;
        bf16_t val = (bf16_t)acc[mi][c].f[r];
        if (mat == 0)
          qb[(size_t)((bb * kNH + h) * kS + s) * kHD + hd] = val;
        else if (mat == 1)
          kb[(size_t)((bb * kNH + h) * kS + s) * kHD + hd] = val;
        else
          vt[(size_t)((bb * kNH + h) * kHD + hd) * kS + s] = val;
      }
    }
  }
}

// ------------------------------------------------- flash attention (causal)
__global__ void __launch_bounds__(256)
k_attn(const bf16_t* __restrict__ qb, const bf16_t* __restrict__ kb,
       const bf16_t* __restrict__ vt, bf16_t* __restrict__ ctxb) {
  __shared__ bf16_t Plds[8][16 * 40];       // per-wave 16x32 P tile, stride 40
  int widb = threadIdx.x >> 5;
  int lane = threadIdx.x & 31;
  int gw   = blockIdx.x * (blockDim.x >> 5) + widb;   // [0, 4096)
  int qt = gw & 127;
  int bh = gw >> 7;                                   // b*16 + h
  int q0 = qt * 16;
  const bf16_t* qbase = qb + (size_t)bh * kS * kHD;
  const bf16_t* kbase = kb + (size_t)bh * kS * kHD;
  const bf16_t* vbase = vt + (size_t)bh * kHD * kS;
  int rA     = lane & 15;
  int aoff   = (lane < 16) ? 0 : 16;
  int boff   = (lane < 16) ? 0 : 16;
  int rowadd = (lane < 16) ? 0 : 8;
  const float scale = 0.08838834764831845f;   // 1/sqrt(128)

  // resident Q A-fragments (16 x 128 -> 4 frags of K=32)
  ABfrag qf[4];
  const bf16_t* qrow = qbase + (size_t)(q0 + rA) * kHD + aoff;
#pragma unroll
  for (int j = 0; j < 4; ++j) load_afrag(qf[j], qrow + j * 32);

  float mrun[8], lrun[8];
  CFrag O[8];
#pragma unroll
  for (int r = 0; r < 8; ++r) { mrun[r] = -3.0e38f; lrun[r] = 0.f; }
#pragma unroll
  for (int c2 = 0; c2 < 8; ++c2)
#pragma unroll
    for (int r = 0; r < 8; ++r) O[c2].f[r] = 0.f;

  bf16_t* Pw = &Plds[widb][0];
  int myrow0 = q0 + rowadd;
  int mycol  = lane & 15;

  for (int k0 = 0; k0 < q0 + 16; k0 += 32) {
    // ---- scores: Q . K^T for 16x32 kv tile (2 C-frags of 16 cols)
    CFrag sc[2];
#pragma unroll
    for (int c = 0; c < 2; ++c)
#pragma unroll
      for (int r = 0; r < 8; ++r) sc[c].f[r] = 0.f;
#pragma unroll
    for (int c = 0; c < 2; ++c) {
      ABfrag kf[4];
      const bf16_t* kp = kbase + (size_t)(k0 + c * 16 + rA) * kHD + boff;
#pragma unroll
      for (int j = 0; j < 4; ++j) load_bfrag(kf[j], kp + j * 32);
#pragma unroll
      for (int j = 0; j < 4; ++j) sc[c].v = wmma_bf16(qf[j].v, kf[j].v, sc[c].v);
    }

    // ---- prefetch first half of V; latency hidden behind softmax VALU work
    ABfrag vf0[4];
#pragma unroll
    for (int c2 = 0; c2 < 4; ++c2)
      load_bfrag(vf0[c2], vbase + (size_t)(c2 * 16 + rA) * kS + k0 + boff);

    // ---- causal mask + online softmax
    float p0v[8], p1v[8];
#pragma unroll
    for (int r = 0; r < 8; ++r) {
      int row = myrow0 + r;
      int c0  = k0 + mycol;
      float v0 = sc[0].f[r] * scale;
      float v1 = sc[1].f[r] * scale;
      v0 = (c0 > row) ? -3.0e38f : v0;
      v1 = (c0 + 16 > row) ? -3.0e38f : v1;
      float tmax = rowred_max16(fmaxf(v0, v1));
      float mo = mrun[r];
      float mn = fmaxf(mo, tmax);
      float corr = __expf(mo - mn);
      float p0 = __expf(v0 - mn);
      float p1 = __expf(v1 - mn);
      float rs = rowred_sum16(p0 + p1);
      lrun[r] = lrun[r] * corr + rs;
      mrun[r] = mn;
#pragma unroll
      for (int c2 = 0; c2 < 8; ++c2) O[c2].f[r] *= corr;
      p0v[r] = p0; p1v[r] = p1;
    }

    // ---- P (C layout) -> LDS -> A layout
#pragma unroll
    for (int r = 0; r < 8; ++r) {
      Pw[(r + rowadd) * 40 + mycol]      = (bf16_t)p0v[r];
      Pw[(r + rowadd) * 40 + 16 + mycol] = (bf16_t)p1v[r];
    }
    asm volatile("s_wait_dscnt 0x0" ::: "memory");
    ABfrag pf;
    load_afrag(pf, Pw + rA * 40 + aoff);

    // ---- second half of V; hidden behind first 4 PV WMMAs
    ABfrag vf1[4];
#pragma unroll
    for (int c2 = 0; c2 < 4; ++c2)
      load_bfrag(vf1[c2], vbase + (size_t)((c2 + 4) * 16 + rA) * kS + k0 + boff);

    // ---- ctx += P . V
#pragma unroll
    for (int c2 = 0; c2 < 4; ++c2)
      O[c2].v = wmma_bf16(pf.v, vf0[c2].v, O[c2].v);
#pragma unroll
    for (int c2 = 0; c2 < 4; ++c2)
      O[c2 + 4].v = wmma_bf16(pf.v, vf1[c2].v, O[c2 + 4].v);
    asm volatile("s_wait_dscnt 0x0" ::: "memory");
  }

  // ---- epilogue: normalize and store ctx in [B,S,D] layout
  float invl[8];
#pragma unroll
  for (int r = 0; r < 8; ++r) invl[r] = 1.0f / lrun[r];
  int b = bh >> 4, h = bh & 15;
#pragma unroll
  for (int c2 = 0; c2 < 8; ++c2) {
    int dcol = h * kHD + c2 * 16 + rA;
#pragma unroll
    for (int r = 0; r < 8; ++r) {
      int s = q0 + r + rowadd;
      ctxb[(size_t)(b * kS + s) * kD + dcol] = (bf16_t)(O[c2].f[r] * invl[r]);
    }
  }
}

// ------------------------------------------- output projection + phase scale
__global__ void __launch_bounds__(256)
k_gemm_out(const bf16_t* __restrict__ ctxb, const bf16_t* __restrict__ Wot,
           const float* __restrict__ phs, float* __restrict__ out) {
  int wid  = (blockIdx.x * blockDim.x + threadIdx.x) >> 5;
  int lane = threadIdx.x & 31;
  int ntile = wid & 31;
  int mtile = wid >> 5;                    // [0, 128)
  int m0 = mtile * 32, n0 = ntile * 64;
  int rA = lane & 15;
  int aoff = (lane < 16) ? 0 : 16;
  int boff = (lane < 16) ? 0 : 16;
  const bf16_t* arow0 = ctxb + (size_t)(m0 + rA) * kD + aoff;
  const bf16_t* arow1 = ctxb + (size_t)(m0 + 16 + rA) * kD + aoff;
  const bf16_t* brow  = Wot + (size_t)(n0 + rA) * kD + boff;

  CFrag acc[2][4];
#pragma unroll
  for (int mi = 0; mi < 2; ++mi)
#pragma unroll
    for (int c = 0; c < 4; ++c)
#pragma unroll
      for (int r = 0; r < 8; ++r) acc[mi][c].f[r] = 0.f;

  ABfrag a0[2], a1[2], b[4];
  load_afrag(a0[0], arow0);
  load_afrag(a0[1], arow1);

  for (int k = 0; k < kD; k += 64) {
#pragma unroll
    for (int c = 0; c < 4; ++c) load_bfrag(b[c], brow + (size_t)c * 16 * kD + k);
    load_afrag(a1[0], arow0 + k + 32);
    load_afrag(a1[1], arow1 + k + 32);
#pragma unroll
    for (int c = 0; c < 4; ++c) {
      acc[0][c].v = wmma_bf16(a0[0].v, b[c].v, acc[0][c].v);
      acc[1][c].v = wmma_bf16(a0[1].v, b[c].v, acc[1][c].v);
    }
#pragma unroll
    for (int c = 0; c < 4; ++c) load_bfrag(b[c], brow + (size_t)c * 16 * kD + k + 32);
    if (k + 64 < kD) {
      load_afrag(a0[0], arow0 + k + 64);
      load_afrag(a0[1], arow1 + k + 64);
    }
#pragma unroll
    for (int c = 0; c < 4; ++c) {
      acc[0][c].v = wmma_bf16(a1[0].v, b[c].v, acc[0][c].v);
      acc[1][c].v = wmma_bf16(a1[1].v, b[c].v, acc[1][c].v);
    }
  }

  int rowadd = (lane < 16) ? 0 : 8;
#pragma unroll
  for (int mi = 0; mi < 2; ++mi) {
#pragma unroll
    for (int c = 0; c < 4; ++c) {
      int n = n0 + c * 16 + rA;
#pragma unroll
      for (int r = 0; r < 8; ++r) {
        int m = m0 + mi * 16 + r + rowadd;
        out[(size_t)m * kD + n] = acc[mi][c].f[r] * (1.0f + 0.1f * phs[m]);
      }
    }
  }
}

// ---------------------------------------------------------------------------
extern "C" void kernel_launch(void* const* d_in, const int* in_sizes, int n_in,
                              void* d_out, int out_size, void* d_ws, size_t ws_size,
                              hipStream_t stream) {
  (void)in_sizes; (void)n_in; (void)out_size; (void)ws_size;
  const float* x   = (const float*)d_in[0];
  const float* Wq  = (const float*)d_in[1];
  const float* Wk  = (const float*)d_in[2];
  const float* Wv  = (const float*)d_in[3];
  const float* Wo  = (const float*)d_in[4];
  const float* Wpq = (const float*)d_in[5];
  const float* Wpk = (const float*)d_in[6];
  float* out = (float*)d_out;

  char* ws = (char*)d_ws;
  const size_t MB = 1024ull * 1024ull;
  bf16_t* xb   = (bf16_t*)(ws + 0);          // 16 MB
  bf16_t* Wqt  = (bf16_t*)(ws + 16 * MB);    // 8 MB each
  bf16_t* Wkt  = (bf16_t*)(ws + 24 * MB);
  bf16_t* Wvt  = (bf16_t*)(ws + 32 * MB);
  bf16_t* Wot  = (bf16_t*)(ws + 40 * MB);
  bf16_t* qbuf = (bf16_t*)(ws + 48 * MB);    // 16 MB each
  bf16_t* kbuf = (bf16_t*)(ws + 64 * MB);
  bf16_t* vtb  = (bf16_t*)(ws + 80 * MB);
  bf16_t* ctxb = (bf16_t*)(ws + 96 * MB);
  float*  cpq  = (float*)(ws + 112 * MB);            // 256 KB each
  float*  spq  = (float*)(ws + 112 * MB + 256 * 1024);
  float*  cpk  = (float*)(ws + 112 * MB + 512 * 1024);
  float*  spk  = (float*)(ws + 112 * MB + 768 * 1024);
  float*  Ck   = (float*)(ws + 113 * MB);
  float*  Sk   = (float*)(ws + 113 * MB + 4096);
  float*  phs  = (float*)(ws + 113 * MB + 8192);

  int nx = kM * kD;                            // 8M elems, 8 per thread
  k_cvt_x<<<nx / (256 * 8), 256, 0, stream>>>(x, xb);
  dim3 gT(kD / 32, kD / 32);
  k_wt<<<gT, 256, 0, stream>>>(Wq, Wqt);
  k_wt<<<gT, 256, 0, stream>>>(Wk, Wkt);
  k_wt<<<gT, 256, 0, stream>>>(Wv, Wvt);
  k_wt<<<gT, 256, 0, stream>>>(Wo, Wot);

  k_phase1<<<kM, 256, 0, stream>>>(x, Wpq, Wpk, cpq, spq, cpk, spk);
  k_phase2<<<kB * kNH, 256, 0, stream>>>(cpk, spk, Ck, Sk);
  k_phase3<<<kM / 256, 256, 0, stream>>>(cpq, spq, Ck, Sk, phs);

  // 3 mats x 128 mtiles x 32 ntiles = 12288 waves -> 1536 blocks of 8 waves
  k_gemm_qkv<<<1536, 256, 0, stream>>>(xb, Wqt, Wkt, Wvt, qbuf, kbuf, vtb);
  // 2*16*128 = 4096 q-tiles -> 512 blocks of 8 waves
  k_attn<<<512, 256, 0, stream>>>(qbuf, kbuf, vtb, ctxb);
  // 128 x 32 = 4096 waves -> 512 blocks
  k_gemm_out<<<512, 256, 0, stream>>>(ctxb, Wot, phs, out);
}
